// CausalSelfAttention_40819369181969
// MI455X (gfx1250) — matrix operand order repaired and checked by reading, not compiled, not run on hardware
//
#include <hip/hip_runtime.h>

typedef __attribute__((ext_vector_type(16))) __bf16 v16bf;
typedef __attribute__((ext_vector_type(8)))  float  v8f;
typedef int v4i __attribute__((vector_size(16)));   // matches async-builtin param type

#define SEQ 4096
#define BATCH 4
#define DMODEL 1024
#define HDIM 128
#define SCALE 0.08838834764831845f  // 1/sqrt(128)

// LDS row strides padded by 8 bf16 (16B) to avoid 64-bank conflicts
#define XSTR 1032   // qkv x-tile row stride (elems)
#define KSTR 136    // flash K block row stride
#define VSTR 72     // flash V block row stride
#define PSTR 72     // flash P tile row stride

#if defined(__AMDGCN__) && __has_builtin(__builtin_amdgcn_global_load_async_to_lds_b128) && \
    __has_builtin(__builtin_amdgcn_s_wait_asynccnt)
#define USE_ASYNC 1
#else
#define USE_ASYNC 0
#endif

union AFrag { v16bf v; uint4 q[2]; };
union BFrag { v16bf v; uint4 q[2]; };

__device__ inline unsigned short f2bf(float f) {
    unsigned int u = __float_as_uint(f);
    u += 0x7FFFu + ((u >> 16) & 1u);   // round-to-nearest-even
    return (unsigned short)(u >> 16);
}

__device__ inline v8f vzero() { v8f z = {0.f,0.f,0.f,0.f,0.f,0.f,0.f,0.f}; return z; }

__device__ inline v8f wmma_bf16(v16bf a, v16bf b, v8f c) {
    return __builtin_amdgcn_wmma_f32_16x16x32_bf16(false, a, false, b, (short)0, c, false, false);
}

__device__ inline float rowmax16(float v) {
    v = fmaxf(v, __shfl_xor(v, 1, 32));
    v = fmaxf(v, __shfl_xor(v, 2, 32));
    v = fmaxf(v, __shfl_xor(v, 4, 32));
    v = fmaxf(v, __shfl_xor(v, 8, 32));
    return v;
}
__device__ inline float rowsum16(float v) {
    v += __shfl_xor(v, 1, 32);
    v += __shfl_xor(v, 2, 32);
    v += __shfl_xor(v, 4, 32);
    v += __shfl_xor(v, 8, 32);
    return v;
}

// 16-byte global -> LDS copy: async DMA (ASYNCcnt) if available, else bounced.
__device__ inline void cp16(const unsigned short* g, unsigned short* l) {
#if USE_ASYNC
    unsigned short* gg = const_cast<unsigned short*>(g);
    __builtin_amdgcn_global_load_async_to_lds_b128(
        (__attribute__((address_space(1))) v4i*)(v4i*)gg,
        (__attribute__((address_space(3))) v4i*)(v4i*)l, 0, 0);
#else
    *(uint4*)l = *(const uint4*)g;
#endif
}

// ---------------- Kernel 0: weight fp32 -> bf16 (each weight = 131072 elems) --
__global__ __launch_bounds__(256) void cvt_weights(
    const float* __restrict__ wq, const float* __restrict__ wk,
    const float* __restrict__ wv, const float* __restrict__ wo,
    unsigned short* __restrict__ Wq, unsigned short* __restrict__ Wk,
    unsigned short* __restrict__ Wv, unsigned short* __restrict__ Wo) {
    int i = blockIdx.x * 256 + threadIdx.x;   // grid = 512 -> 131072 threads
    Wq[i] = f2bf(wq[i]);
    Wk[i] = f2bf(wk[i]);
    Wv[i] = f2bf(wv[i]);
    Wo[i] = f2bf(wo[i]);
}

// ---------------- Kernel 1: fused QKV projection ------------------------------
// x: (S,B,D) fp32, row r = s*4+b.  Q,K out: (B,S,H) bf16.  V out: (B,H,S) bf16.
__global__ __launch_bounds__(256) void qkv_proj(
    const float* __restrict__ x,
    const unsigned short* __restrict__ Wq, const unsigned short* __restrict__ Wk,
    const unsigned short* __restrict__ Wv,
    unsigned short* __restrict__ Qb, unsigned short* __restrict__ Kb,
    unsigned short* __restrict__ Vt) {
    __shared__ __align__(16) unsigned short xs[16 * XSTR];  // padded bf16 tile

    const int tile = blockIdx.x;            // 16-row tile of R = S*B = 16384
    const int tid  = threadIdx.x;
    const float* xrow = x + (size_t)tile * 16 * DMODEL;
#pragma unroll
    for (int i = 0; i < 64; ++i) {
        int idx = tid + i * 256;
        int r = idx >> 10, cc = idx & 1023;
        xs[r * XSTR + cc] = f2bf(xrow[idx]);
    }
    __syncthreads();

    const int w = tid >> 5, lane = tid & 31, half = lane >> 4, l16 = lane & 15;
    const int h0 = w * 16;

    v8f cq = vzero(), ck = vzero(), cv = vzero();
    for (int kc = 0; kc < 32; ++kc) {
        const int k0 = kc * 32;
        AFrag a;
        a.q[0] = *(const uint4*)(xs + l16 * XSTR + k0 + half * 16);
        a.q[1] = *(const uint4*)(xs + l16 * XSTR + k0 + half * 16 + 8);
        BFrag bq, bk, bv;
        const unsigned short* pq = Wq + (size_t)(h0 + l16) * DMODEL + k0 + half * 16;
        const unsigned short* pk = Wk + (size_t)(h0 + l16) * DMODEL + k0 + half * 16;
        const unsigned short* pv = Wv + (size_t)(h0 + l16) * DMODEL + k0 + half * 16;
        bq.q[0] = *(const uint4*)pq;  bq.q[1] = *(const uint4*)(pq + 8);
        bk.q[0] = *(const uint4*)pk;  bk.q[1] = *(const uint4*)(pk + 8);
        bv.q[0] = *(const uint4*)pv;  bv.q[1] = *(const uint4*)(pv + 8);
        cq = wmma_bf16(a.v, bq.v, cq);
        ck = wmma_bf16(a.v, bk.v, ck);
        cv = wmma_bf16(a.v, bv.v, cv);
    }
#pragma unroll
    for (int i = 0; i < 8; ++i) {
        int r = tile * 16 + half * 8 + i;
        int s = r >> 2, bb = r & 3;
        int h = h0 + l16;
        Qb[((size_t)(bb * SEQ + s)) * HDIM + h] = f2bf(cq[i]);
        Kb[((size_t)(bb * SEQ + s)) * HDIM + h] = f2bf(ck[i]);
        Vt[((size_t)(bb * HDIM + h)) * SEQ + s] = f2bf(cv[i]);
    }
}

// Cooperative staging of one 64-key K/V block into (padded) LDS.
// K: 64 x 128, thread copies half a row. V (transposed): 128 x 64, thread copies one row.
__device__ inline void stage_kv(const unsigned short* __restrict__ Kb,
                                const unsigned short* __restrict__ Vt,
                                unsigned short* Ksd, unsigned short* Vsd,
                                int b, int kk, int tid) {
    {
        const int r = tid >> 1, c0 = (tid & 1) * 64;
        const unsigned short* src = Kb + ((size_t)(b * SEQ + kk + r)) * HDIM + c0;
        unsigned short* dst = Ksd + r * KSTR + c0;
#pragma unroll
        for (int j = 0; j < 8; ++j) cp16(src + j * 8, dst + j * 8);
    }
    {
        const unsigned short* src = Vt + ((size_t)(b * HDIM + tid)) * SEQ + kk;
        unsigned short* dst = Vsd + tid * VSTR;
#pragma unroll
        for (int j = 0; j < 8; ++j) cp16(src + j * 8, dst + j * 8);
    }
}

// ---------------- Kernel 2: causal flash attention ----------------------------
// grid (64, 4): blockIdx.x = 64-row q block, blockIdx.y = batch. 128 thr = 4 waves.
// Key-block count (q0+79)>>6 == blockIdx.x+1 is uniform across the 4 waves, so
// cooperative double-buffered staging is divergence-free.
__global__ __launch_bounds__(128) void flash_attn(
    const unsigned short* __restrict__ Qb, const unsigned short* __restrict__ Kb,
    const unsigned short* __restrict__ Vt, unsigned short* __restrict__ Ob) {
    __shared__ __align__(16) unsigned short Ks[2][64 * KSTR];
    __shared__ __align__(16) unsigned short Vs[2][128 * VSTR];
    __shared__ __align__(16) unsigned short Pb[4][16 * PSTR];

    const int b   = blockIdx.y;
    const int tid = threadIdx.x;
    const int w = tid >> 5, lane = tid & 31, half = lane >> 4, l16 = lane & 15;
    const int q0 = blockIdx.x * 64 + w * 16;

    // Q A-fragments for this wave's 16 q-rows (K dim = H = 128 -> 4 chunks)
    AFrag aQ[4];
    {
        const unsigned short* qr = Qb + ((size_t)(b * SEQ + q0 + l16)) * HDIM;
#pragma unroll
        for (int c = 0; c < 4; ++c) {
            aQ[c].q[0] = *(const uint4*)(qr + c * 32 + half * 16);
            aQ[c].q[1] = *(const uint4*)(qr + c * 32 + half * 16 + 8);
        }
    }

    v8f acc[8];
    float m_run[8], l_run[8];
#pragma unroll
    for (int t = 0; t < 8; ++t) acc[t] = vzero();
#pragma unroll
    for (int i = 0; i < 8; ++i) { m_run[i] = -3.0e38f; l_run[i] = 0.f; }

    const int nkb = blockIdx.x + 1;
    stage_kv(Kb, Vt, Ks[0], Vs[0], b, 0, tid);   // prologue

    for (int kb = 0; kb < nkb; ++kb) {
        const int kk = kb * 64;
#if USE_ASYNC
        __builtin_amdgcn_s_wait_asynccnt(0);
#endif
        __syncthreads();   // buffer kb ready; buffer kb-1 free for overwrite
        if (kb + 1 < nkb)
            stage_kv(Kb, Vt, Ks[(kb + 1) & 1], Vs[(kb + 1) & 1], b, kk + 64, tid);

        const unsigned short* Ksp = Ks[kb & 1];
        const unsigned short* Vsp = Vs[kb & 1];

        // ---- scores S = Q K^T (4 key tiles x 4 h-chunks) ----
        v8f sc[4];
#pragma unroll
        for (int t = 0; t < 4; ++t) sc[t] = vzero();
#pragma unroll
        for (int c = 0; c < 4; ++c) {
#pragma unroll
            for (int t = 0; t < 4; ++t) {
                BFrag bK;
                const unsigned short* kp = Ksp + (t * 16 + l16) * KSTR + c * 32 + half * 16;
                bK.q[0] = *(const uint4*)kp;
                bK.q[1] = *(const uint4*)(kp + 8);
                sc[t] = wmma_bf16(aQ[c].v, bK.v, sc[t]);
            }
        }
        // ---- scale + causal mask (C layout: col = l16, row = half*8+i) ----
#pragma unroll
        for (int t = 0; t < 4; ++t) {
            const int kcol = kk + t * 16 + l16;
#pragma unroll
            for (int i = 0; i < 8; ++i) {
                const int qrow = q0 + half * 8 + i;
                float v = sc[t][i] * SCALE;
                sc[t][i] = (kcol <= qrow) ? v : -3.0e38f;
            }
        }
        // ---- online softmax update ----
        float alpha[8];
#pragma unroll
        for (int i = 0; i < 8; ++i) {
            float mx = fmaxf(fmaxf(sc[0][i], sc[1][i]), fmaxf(sc[2][i], sc[3][i]));
            mx = rowmax16(mx);
            float mn = fmaxf(m_run[i], mx);
            alpha[i] = __expf(m_run[i] - mn);
            m_run[i] = mn;
            float s = 0.f;
#pragma unroll
            for (int t = 0; t < 4; ++t) {
                float p = __expf(sc[t][i] - mn);
                sc[t][i] = p;
                s += p;
            }
            s = rowsum16(s);
            l_run[i] = l_run[i] * alpha[i] + s;
        }
#pragma unroll
        for (int t = 0; t < 8; ++t)
#pragma unroll
            for (int i = 0; i < 8; ++i) acc[t][i] *= alpha[i];
        // ---- P: C-layout -> LDS -> A-layout (same-wave DS ops are in order) ----
        unsigned short* pb = &Pb[w][0];
#pragma unroll
        for (int t = 0; t < 4; ++t)
#pragma unroll
            for (int i = 0; i < 8; ++i)
                pb[(half * 8 + i) * PSTR + t * 16 + l16] = f2bf(sc[t][i]);
        AFrag aP[2];
#pragma unroll
        for (int c = 0; c < 2; ++c) {
            aP[c].q[0] = *(const uint4*)(pb + l16 * PSTR + c * 32 + half * 16);
            aP[c].q[1] = *(const uint4*)(pb + l16 * PSTR + c * 32 + half * 16 + 8);
        }
        // ---- acc += P V : c outer so 8 consecutive WMMAs share the A operand ----
#pragma unroll
        for (int c = 0; c < 2; ++c) {
#pragma unroll
            for (int t8 = 0; t8 < 8; ++t8) {
                BFrag bV;
                const unsigned short* vp = Vsp + (t8 * 16 + l16) * VSTR + c * 32 + half * 16;
                bV.q[0] = *(const uint4*)vp;
                bV.q[1] = *(const uint4*)(vp + 8);
                acc[t8] = wmma_bf16(aP[c].v, bV.v, acc[t8]);
            }
        }
    }
    // ---- epilogue: normalize, store O (B,S,H) bf16 ----
    float inv[8];
#pragma unroll
    for (int i = 0; i < 8; ++i) inv[i] = 1.0f / l_run[i];
#pragma unroll
    for (int t8 = 0; t8 < 8; ++t8)
#pragma unroll
        for (int i = 0; i < 8; ++i)
            Ob[((size_t)(b * SEQ + q0 + half * 8 + i)) * HDIM + t8 * 16 + l16] =
                f2bf(acc[t8][i] * inv[i]);
}

// ---------------- Kernel 3: output projection y[s,b,d] = O[b,s,:] . Wo[d,:] --
__global__ __launch_bounds__(256) void out_proj(
    const unsigned short* __restrict__ Ob, const unsigned short* __restrict__ Wo,
    float* __restrict__ y) {
    const int tile = blockIdx.x;  // 16-row tile of R = 16384 (r = s*4+b)
    const int tid = threadIdx.x;
    const int w = tid >> 5, lane = tid & 31, half = lane >> 4, l16 = lane & 15;

    AFrag a[4];
    {
        int r = tile * 16 + l16;
        int s = r >> 2, bb = r & 3;
        const unsigned short* orow = Ob + ((size_t)(bb * SEQ + s)) * HDIM;
#pragma unroll
        for (int c = 0; c < 4; ++c) {
            a[c].q[0] = *(const uint4*)(orow + c * 32 + half * 16);
            a[c].q[1] = *(const uint4*)(orow + c * 32 + half * 16 + 8);
        }
    }
#pragma unroll
    for (int j = 0; j < 8; ++j) {
        const int d0 = w * 128 + j * 16;
        v8f c = vzero();
#pragma unroll
        for (int kc = 0; kc < 4; ++kc) {
            BFrag bw;
            const unsigned short* wp = Wo + (size_t)(d0 + l16) * HDIM + kc * 32 + half * 16;
            bw.q[0] = *(const uint4*)wp;
            bw.q[1] = *(const uint4*)(wp + 8);
            c = wmma_bf16(a[kc].v, bw.v, c);
        }
#pragma unroll
        for (int i = 0; i < 8; ++i) {
            int r = tile * 16 + half * 8 + i;
            y[(size_t)r * DMODEL + d0 + l16] = c[i];
        }
    }
}

// -----------------------------------------------------------------------------
extern "C" void kernel_launch(void* const* d_in, const int* in_sizes, int n_in,
                              void* d_out, int out_size, void* d_ws, size_t ws_size,
                              hipStream_t stream) {
    const float* x  = (const float*)d_in[0];
    const float* wq = (const float*)d_in[1];
    const float* wk = (const float*)d_in[2];
    const float* wv = (const float*)d_in[3];
    const float* wo = (const float*)d_in[4];

    const size_t NBSH = (size_t)BATCH * SEQ * HDIM;   // 2M elems
    const size_t NW   = (size_t)HDIM * DMODEL;        // 131072 elems
    unsigned short* Qb  = (unsigned short*)d_ws;
    unsigned short* Kb  = Qb + NBSH;
    unsigned short* Vt  = Kb + NBSH;
    unsigned short* Ob  = Vt + NBSH;
    unsigned short* Wqb = Ob + NBSH;
    unsigned short* Wkb = Wqb + NW;
    unsigned short* Wvb = Wkb + NW;
    unsigned short* Wob = Wvb + NW;

    cvt_weights<<<512, 256, 0, stream>>>(wq, wk, wv, wo, Wqb, Wkb, Wvb, Wob);
    qkv_proj<<<(SEQ * BATCH) / 16, 256, 0, stream>>>(x, Wqb, Wkb, Wvb, Qb, Kb, Vt);
    flash_attn<<<dim3(SEQ / 64, BATCH), 128, 0, stream>>>(Qb, Kb, Vt, Ob);
    out_proj<<<(SEQ * BATCH) / 16, 256, 0, stream>>>(Ob, Wob, (float*)d_out);
}
